// TransformerBlock_58317065945646
// MI455X (gfx1250) — matrix operand order repaired and checked by reading, not compiled, not run on hardware
//
#include <hip/hip_runtime.h>
#include <hip/hip_bf16.h>
#include <cstdint>

typedef unsigned short u16;
typedef __bf16 bf16_t;
typedef bf16_t v16bf __attribute__((ext_vector_type(16)));
typedef float  v8f   __attribute__((ext_vector_type(8)));

union AFrag { uint4 q[2]; v16bf v; };
union BFrag { uint4 q[2]; v16bf v; };
union U4H   { uint4 u; u16 h[8]; };

__device__ __forceinline__ u16 f2bf(float f) {
    union { float f; unsigned int u; } x; x.f = f;
    unsigned int lsb = (x.u >> 16) & 1u;
    unsigned int r = x.u + 0x7FFFu + lsb;
    return (u16)(r >> 16);
}

// CDNA5 async global->LDS copy (16B per lane), tracked by ASYNCcnt.
__device__ __forceinline__ void async_copy_b128(const u16* gptr, u16* lptr) {
    uint32_t lds = (uint32_t)(uintptr_t)lptr;  // low 32 bits of generic ptr = LDS offset
    asm volatile("global_load_async_to_lds_b128 %0, %1, off"
                 :: "v"(lds), "v"(gptr) : "memory");
}
__device__ __forceinline__ void wait_async0() {
    asm volatile("s_wait_asynccnt 0x0" ::: "memory");
}

// ---------------------------------------------------------------------------
// fp32 -> bf16 conversion (weights)
// ---------------------------------------------------------------------------
__global__ void f32_to_bf16_kernel(const float* __restrict__ in, u16* __restrict__ out, int n) {
    for (int i = blockIdx.x * blockDim.x + threadIdx.x; i < n; i += gridDim.x * blockDim.x)
        out[i] = f2bf(in[i]);
}

// ---------------------------------------------------------------------------
// LayerNorm (row of 512) fused with bf16 conversion
// ---------------------------------------------------------------------------
__global__ __launch_bounds__(256) void layernorm_bf16_kernel(
    const float* __restrict__ x, const float* __restrict__ g, const float* __restrict__ b,
    u16* __restrict__ out)
{
    const int D = 512;
    int row = blockIdx.x;
    const float* xr = x + (size_t)row * D;
    float v0 = xr[threadIdx.x], v1 = xr[threadIdx.x + 256];
    float s = v0 + v1, s2 = v0 * v0 + v1 * v1;
    for (int off = 16; off >= 1; off >>= 1) {
        s  += __shfl_xor(s,  off, 32);
        s2 += __shfl_xor(s2, off, 32);
    }
    __shared__ float red0[8], red1[8];
    int wid = threadIdx.x >> 5, lid = threadIdx.x & 31;
    if (lid == 0) { red0[wid] = s; red1[wid] = s2; }
    __syncthreads();
    s = 0.f; s2 = 0.f;
    for (int w = 0; w < 8; ++w) { s += red0[w]; s2 += red1[w]; }
    float mu   = s * (1.f / D);
    float var  = s2 * (1.f / D) - mu * mu;
    float rstd = rsqrtf(var + 1e-5f);
    int i0 = threadIdx.x, i1 = threadIdx.x + 256;
    out[(size_t)row * D + i0] = f2bf((v0 - mu) * rstd * g[i0] + b[i0]);
    out[(size_t)row * D + i1] = f2bf((v1 - mu) * rstd * g[i1] + b[i1]);
}

// ---------------------------------------------------------------------------
// Tiled bf16 WMMA GEMM with double-buffered LDS:
//   A tile (128x32) staged via global_load_async_to_lds_b128 (no transpose),
//   B tile (32x128) staged via register transpose. One barrier per K-step.
// 8 waves (2x4), each wave: 4x2 fragments of v_wmma_f32_16x16x32_bf16.
// ---------------------------------------------------------------------------
#define LDA 40  // 32 + 8 pad (16B-aligned rows, spreads LDS banks)

__device__ __forceinline__ void gemm_stageA_async(
    const u16* __restrict__ A, u16* dst, int m0, int K, int k0, int tid)
{
#pragma unroll
    for (int it = 0; it < 2; ++it) {
        int idx = tid + it * 256;           // 0..511
        int r = idx >> 2, q = idx & 3;      // 128 rows x 4 quads
        async_copy_b128(A + (size_t)(m0 + r) * K + k0 + q * 8, dst + r * LDA + q * 8);
    }
}
__device__ __forceinline__ void gemm_loadB(
    const u16* __restrict__ Bw, int Nc, int n0, int k0, int tid, uint4* reg)
{
#pragma unroll
    for (int it = 0; it < 2; ++it) {
        int idx = tid + it * 256;
        int r = idx >> 4, q = idx & 15;     // 32 k-rows x 16 quads
        reg[it] = *reinterpret_cast<const uint4*>(Bw + (size_t)(k0 + r) * Nc + n0 + q * 8);
    }
}
__device__ __forceinline__ void gemm_storeB(u16* dst, int tid, const uint4* reg)
{
#pragma unroll
    for (int it = 0; it < 2; ++it) {
        int idx = tid + it * 256;
        int r = idx >> 4, q = idx & 15;
        U4H t; t.u = reg[it];
#pragma unroll
        for (int e = 0; e < 8; ++e) dst[(q * 8 + e) * LDA + r] = t.h[e];
    }
}

__global__ __launch_bounds__(256) void gemm_bf16_wmma(
    const u16* __restrict__ A, const u16* __restrict__ Bw,
    const float* __restrict__ bias, const float* __restrict__ residual,
    float* __restrict__ outF, u16* __restrict__ outB,
    int M, int K, int Nc, int doGelu)
{
    __shared__ u16 As[2][128 * LDA];
    __shared__ u16 Bs[2][128 * LDA];

    int tid  = threadIdx.x;
    int wave = tid >> 5, lane = tid & 31;
    int hi   = (lane >= 16) ? 1 : 0;
    int l16  = lane & 15;
    int m0 = blockIdx.y * 128, n0 = blockIdx.x * 128;
    int wm = (wave >> 2) * 64, wn = (wave & 3) * 32;
    int ko = hi ? 8 : 0, kb = hi ? 16 : 0;

    v8f acc[4][2];
    v8f zero8 = {0.f, 0.f, 0.f, 0.f, 0.f, 0.f, 0.f, 0.f};
#pragma unroll
    for (int i = 0; i < 4; ++i)
#pragma unroll
        for (int j = 0; j < 2; ++j) acc[i][j] = zero8;

    // Prologue: stage tile 0
    uint4 breg[2];
    gemm_stageA_async(A, As[0], m0, K, 0, tid);
    gemm_loadB(Bw, Nc, n0, 0, tid, breg);
    gemm_storeB(Bs[0], tid, breg);
    wait_async0();
    __syncthreads();

    int buf = 0;
    for (int k0 = 0; k0 < K; k0 += 32) {
        int nxt = buf ^ 1;
        bool more = (k0 + 32) < K;
        if (more) {
            gemm_stageA_async(A, As[nxt], m0, K, k0 + 32, tid);
            gemm_loadB(Bw, Nc, n0, k0 + 32, tid, breg);
        }

        AFrag af[4];
        BFrag bfr[2];
#pragma unroll
        for (int i = 0; i < 4; ++i) {
            int row = wm + i * 16 + l16;
            af[i].q[0] = *reinterpret_cast<const uint4*>(&As[buf][row * LDA + ko]);
            af[i].q[1] = *reinterpret_cast<const uint4*>(&As[buf][row * LDA + ko + 16]);
        }
#pragma unroll
        for (int j = 0; j < 2; ++j) {
            int col = wn + j * 16 + l16;
            bfr[j].q[0] = *reinterpret_cast<const uint4*>(&Bs[buf][col * LDA + ko]);
            bfr[j].q[1] = *reinterpret_cast<const uint4*>(&Bs[buf][col * LDA + ko + 16]);
        }
#pragma unroll
        for (int i = 0; i < 4; ++i)
#pragma unroll
            for (int j = 0; j < 2; ++j)
                acc[i][j] = __builtin_amdgcn_wmma_f32_16x16x32_bf16(
                    false, af[i].v, false, bfr[j].v, (short)0, acc[i][j], false, false);

        if (more) {
            gemm_storeB(Bs[nxt], tid, breg);
            wait_async0();
        }
        __syncthreads();
        buf = nxt;
    }

    // Epilogue: bias, optional GELU, optional residual, fp32 and/or bf16 out
#pragma unroll
    for (int i = 0; i < 4; ++i)
#pragma unroll
        for (int j = 0; j < 2; ++j)
#pragma unroll
            for (int v = 0; v < 8; ++v) {
                int gr = m0 + wm + i * 16 + v + 8 * hi;
                int gc = n0 + wn + j * 16 + l16;
                float val = acc[i][j][v] + bias[gc];
                if (doGelu) val = 0.5f * val * (1.f + erff(val * 0.70710678118f));
                if (residual) val += residual[(size_t)gr * Nc + gc];
                if (outF) outF[(size_t)gr * Nc + gc] = val;
                if (outB) outB[(size_t)gr * Nc + gc] = f2bf(val);
            }
}

// ---------------------------------------------------------------------------
// Flash attention, double-buffered 32-key chunks:
//   K chunk staged async (row-major), V chunk register-transposed.
//   One block barrier per chunk; P relayout is wave-private (DScnt-ordered).
// ---------------------------------------------------------------------------
#define LDK 72  // 64 + 8 pad
#define LDV 40  // 32 + 8 pad

__device__ __forceinline__ void attn_stageK_async(
    const u16* __restrict__ qkv, u16* dst, size_t rowBase, int j0, int h, int tid)
{
#pragma unroll
    for (int it = 0; it < 2; ++it) {
        int idx = tid + it * 128;           // 0..255
        int r = idx >> 3, q = idx & 7;      // 32 keys x 8 quads
        async_copy_b128(qkv + (rowBase + j0 + r) * 1536 + 512 + h * 64 + q * 8,
                        dst + r * LDK + q * 8);
    }
}
__device__ __forceinline__ void attn_loadV(
    const u16* __restrict__ qkv, size_t rowBase, int j0, int h, int tid, uint4* reg)
{
#pragma unroll
    for (int it = 0; it < 2; ++it) {
        int idx = tid + it * 128;
        int r = idx >> 3, q = idx & 7;
        reg[it] = *reinterpret_cast<const uint4*>(
            qkv + (rowBase + j0 + r) * 1536 + 1024 + h * 64 + q * 8);
    }
}
__device__ __forceinline__ void attn_storeVt(u16* dst, int tid, const uint4* reg)
{
#pragma unroll
    for (int it = 0; it < 2; ++it) {
        int idx = tid + it * 128;
        int r = idx >> 3, q = idx & 7;
        U4H t; t.u = reg[it];
#pragma unroll
        for (int e = 0; e < 8; ++e) dst[(q * 8 + e) * LDV + r] = t.h[e];
    }
}

__global__ __launch_bounds__(128) void attention_kernel(
    const u16* __restrict__ qkv, u16* __restrict__ attn)
{
    const int Nn = 2048;
    __shared__ u16 Ks[2][32 * LDK];
    __shared__ u16 Vt[2][64 * LDV];
    __shared__ u16 Pl[4][16 * LDV];

    int tid = threadIdx.x, wave = tid >> 5, lane = tid & 31;
    int hi = (lane >= 16) ? 1 : 0, l16 = lane & 15;
    int bh = blockIdx.y;
    int b = bh >> 3, h = bh & 7;
    int n0 = blockIdx.x * 64 + wave * 16;
    size_t rowBase = (size_t)b * Nn;
    int ko = hi ? 8 : 0, kb = hi ? 16 : 0;

    // Q fragments (DH=64 -> two 32-wide K-steps), loaded once
    AFrag qf[2];
    {
        int qrow = n0 + l16;
        const u16* p = qkv + (rowBase + qrow) * 1536 + h * 64;
#pragma unroll
        for (int st = 0; st < 2; ++st) {
            qf[st].q[0] = *reinterpret_cast<const uint4*>(p + st * 32 + ko);
            qf[st].q[1] = *reinterpret_cast<const uint4*>(p + st * 32 + ko + 16);
        }
    }

    v8f zero8 = {0.f, 0.f, 0.f, 0.f, 0.f, 0.f, 0.f, 0.f};
    v8f o[4];
#pragma unroll
    for (int f = 0; f < 4; ++f) o[f] = zero8;
    float mi[8], li[8];
#pragma unroll
    for (int v = 0; v < 8; ++v) { mi[v] = -1e30f; li[v] = 0.f; }

    u16* prow = &Pl[wave][0];

    // Prologue: stage chunk 0
    uint4 vreg[2];
    attn_stageK_async(qkv, Ks[0], rowBase, 0, h, tid);
    attn_loadV(qkv, rowBase, 0, h, tid, vreg);
    attn_storeVt(Vt[0], tid, vreg);
    wait_async0();
    __syncthreads();

    int buf = 0;
    for (int j0 = 0; j0 < Nn; j0 += 32) {
        int nxt = buf ^ 1;
        bool more = (j0 + 32) < Nn;
        if (more) {
            attn_stageK_async(qkv, Ks[nxt], rowBase, j0 + 32, h, tid);
            attn_loadV(qkv, rowBase, j0 + 32, h, tid, vreg);
        }

        // S(16x32) = Q @ K^T
        v8f s[2]; s[0] = zero8; s[1] = zero8;
#pragma unroll
        for (int nf = 0; nf < 2; ++nf) {
            int key = nf * 16 + l16;
#pragma unroll
            for (int st = 0; st < 2; ++st) {
                BFrag kf;
                const u16* p = &Ks[buf][key * LDK + st * 32 + ko];
                kf.q[0] = *reinterpret_cast<const uint4*>(p);
                kf.q[1] = *reinterpret_cast<const uint4*>(p + 16);
                s[nf] = __builtin_amdgcn_wmma_f32_16x16x32_bf16(
                    false, qf[st].v, false, kf.v, (short)0, s[nf], false, false);
            }
        }

        // Online softmax over 32 keys
        float corr[8];
#pragma unroll
        for (int v = 0; v < 8; ++v) {
            float v0 = s[0][v] * 0.125f;  // DH^-0.5
            float v1 = s[1][v] * 0.125f;
            float mx = fmaxf(v0, v1);
            for (int off = 8; off >= 1; off >>= 1) mx = fmaxf(mx, __shfl_xor(mx, off, 16));
            float mnew = fmaxf(mi[v], mx);
            float p0 = __expf(v0 - mnew), p1 = __expf(v1 - mnew);
            float rs = p0 + p1;
            for (int off = 8; off >= 1; off >>= 1) rs += __shfl_xor(rs, off, 16);
            float c = __expf(mi[v] - mnew);
            li[v] = li[v] * c + rs;
            mi[v] = mnew;
            corr[v] = c;
            int r = v + 8 * hi;
            prow[r * LDV + l16]      = f2bf(p0);
            prow[r * LDV + 16 + l16] = f2bf(p1);
        }
#pragma unroll
        for (int f = 0; f < 4; ++f)
#pragma unroll
            for (int v = 0; v < 8; ++v) o[f][v] *= corr[v];

        // Wave-private LDS relayout of P (same-wave DS RAW, ordered by DScnt)
        __builtin_amdgcn_wave_barrier();
        AFrag pf;
        {
            const u16* p = &prow[l16 * LDV + ko];
            pf.q[0] = *reinterpret_cast<const uint4*>(p);
            pf.q[1] = *reinterpret_cast<const uint4*>(p + 16);
        }
        // O += P @ V
#pragma unroll
        for (int f = 0; f < 4; ++f) {
            BFrag vf;
            const u16* p = &Vt[buf][(f * 16 + l16) * LDV + ko];
            vf.q[0] = *reinterpret_cast<const uint4*>(p);
            vf.q[1] = *reinterpret_cast<const uint4*>(p + 16);
            o[f] = __builtin_amdgcn_wmma_f32_16x16x32_bf16(
                false, pf.v, false, vf.v, (short)0, o[f], false, false);
        }

        if (more) {
            attn_storeVt(Vt[nxt], tid, vreg);
            wait_async0();
        }
        __syncthreads();
        buf = nxt;
    }

    // Normalize and store bf16 [B*N, D]
#pragma unroll
    for (int f = 0; f < 4; ++f)
#pragma unroll
        for (int v = 0; v < 8; ++v) {
            float val = o[f][v] / li[v];
            int gr = (int)rowBase + n0 + v + 8 * hi;
            int gc = h * 64 + f * 16 + l16;
            attn[(size_t)gr * 512 + gc] = f2bf(val);
        }
}

// ---------------------------------------------------------------------------
// Host launcher
// ---------------------------------------------------------------------------
extern "C" void kernel_launch(void* const* d_in, const int* in_sizes, int n_in,
                              void* d_out, int out_size, void* d_ws, size_t ws_size,
                              hipStream_t stream) {
    (void)in_sizes; (void)n_in; (void)out_size; (void)ws_size;
    const int B = 4, N = 2048, D = 512, DFF = 2048;
    const int M = B * N;  // 8192

    const float* x     = (const float*)d_in[0];
    const float* ln1_g = (const float*)d_in[1];
    const float* ln1_b = (const float*)d_in[2];
    const float* Wqkv  = (const float*)d_in[3];
    const float* bqkv  = (const float*)d_in[4];
    const float* Wproj = (const float*)d_in[5];
    const float* bproj = (const float*)d_in[6];
    const float* ln2_g = (const float*)d_in[7];
    const float* ln2_b = (const float*)d_in[8];
    const float* W1    = (const float*)d_in[9];
    const float* b1    = (const float*)d_in[10];
    const float* W2    = (const float*)d_in[11];
    const float* b2    = (const float*)d_in[12];
    float* out = (float*)d_out;

    char* ws = (char*)d_ws;
    size_t off = 0;
    auto alloc = [&](size_t bytes) -> void* {
        void* p = ws + off;
        off = (off + bytes + 255) & ~(size_t)255;
        return p;
    };
    u16* Wqkv_bf = (u16*)alloc((size_t)D * 3 * D * 2);
    u16* Wproj_bf= (u16*)alloc((size_t)D * D * 2);
    u16* W1_bf   = (u16*)alloc((size_t)D * DFF * 2);
    u16* W2_bf   = (u16*)alloc((size_t)DFF * D * 2);
    u16* h_bf    = (u16*)alloc((size_t)M * D * 2);       // LN1 out, reused for LN2 out
    u16* qkv_bf  = (u16*)alloc((size_t)M * 3 * D * 2);
    u16* attn_bf = (u16*)alloc((size_t)M * D * 2);
    float* x2    = (float*)alloc((size_t)M * D * 4);
    u16* a1_bf   = (u16*)alloc((size_t)M * DFF * 2);

    // Weight conversions
    f32_to_bf16_kernel<<<1024, 256, 0, stream>>>(Wqkv,  Wqkv_bf,  D * 3 * D);
    f32_to_bf16_kernel<<<1024, 256, 0, stream>>>(Wproj, Wproj_bf, D * D);
    f32_to_bf16_kernel<<<1024, 256, 0, stream>>>(W1,    W1_bf,    D * DFF);
    f32_to_bf16_kernel<<<1024, 256, 0, stream>>>(W2,    W2_bf,    DFF * D);

    // LN1
    layernorm_bf16_kernel<<<M, 256, 0, stream>>>(x, ln1_g, ln1_b, h_bf);

    // QKV GEMM: [8192,512] @ [512,1536] -> bf16
    gemm_bf16_wmma<<<dim3((3 * D) / 128, M / 128), 256, 0, stream>>>(
        h_bf, Wqkv_bf, bqkv, nullptr, nullptr, qkv_bf, M, D, 3 * D, 0);

    // Attention
    attention_kernel<<<dim3(N / 64, B * 8), 128, 0, stream>>>(qkv_bf, attn_bf);

    // Proj GEMM + residual(x): -> x2 (fp32)
    gemm_bf16_wmma<<<dim3(D / 128, M / 128), 256, 0, stream>>>(
        attn_bf, Wproj_bf, bproj, x, x2, nullptr, M, D, D, 0);

    // LN2
    layernorm_bf16_kernel<<<M, 256, 0, stream>>>(x2, ln2_g, ln2_b, h_bf);

    // FFN1 GEMM + GELU: [8192,512] @ [512,2048] -> bf16
    gemm_bf16_wmma<<<dim3(DFF / 128, M / 128), 256, 0, stream>>>(
        h_bf, W1_bf, b1, nullptr, nullptr, a1_bf, M, D, DFF, 1);

    // FFN2 GEMM + residual(x2): [8192,2048] @ [2048,512] -> d_out (fp32)
    gemm_bf16_wmma<<<dim3(D / 128, M / 128), 256, 0, stream>>>(
        a1_bf, W2_bf, b2, x2, out, nullptr, M, DFF, D, 0);
}
